// RagatConvE_70196945486559
// MI455X (gfx1250) — compile-verified, matrix operand order checked by reading, not run on hardware
//
#include <hip/hip_runtime.h>
#include <cstdint>

// ---------------------------------------------------------------------------
// RAGAT + ConvE forward for MI455X (gfx1250, wave32, WMMA).
// All matrix math runs through v_wmma_f32_16x16x32_bf16 (bf16 in, f32 acc).
// GEMM: 2x2 tiles (32x32) per wave, K templated (compile-time trip count),
// software-pipelined fragment loads with branchless steady state.
// Requires ws_size >= ~540 MB.
// ---------------------------------------------------------------------------

typedef unsigned short u16;
typedef unsigned int   u32;

typedef __attribute__((ext_vector_type(16))) __bf16 v16bf;
typedef __attribute__((ext_vector_type(8)))  float  v8f;

#define DEVFN __device__ __forceinline__

#define NUM_ENT   40000
#define NUM_REL2  400        // 2*NUM_REL
#define E_TOT     640000
#define E_HALF    320000
#define BSZ       1024
#define P100      128        // padded K for dim-100 (mult of 32)
#define P200      224        // padded feature dim for 200 (mult of 32 and 16)
#define PCONV     208        // padded filter count (mult of 16)
#define KCONV     64         // padded 7*7=49 (mult of 32)
#define FLATSZ    39200      // 14*14*200, already mult of 32
#define EPS_BN    1e-5f

// ---------------- scalar bf16 helpers (storage as u16) ---------------------
DEVFN u16 f2bf(float f) {
    u32 u = __float_as_uint(f);
    u32 r = u + 0x7FFFu + ((u >> 16) & 1u);   // round-to-nearest-even
    return (u16)(r >> 16);
}
DEVFN float bf2f(u16 h) { return __uint_as_float(((u32)h) << 16); }

// monotone float<->u32 map for atomicMax-based float max
DEVFN u32 fenc(float f) {
    u32 u = __float_as_uint(f);
    return (u & 0x80000000u) ? ~u : (u | 0x80000000u);
}
DEVFN float fdec(u32 u) {
    u32 b = (u & 0x80000000u) ? (u & 0x7FFFFFFFu) : ~u;
    return __uint_as_float(b);
}
#define ENC_NEG_INF 0x007FFFFFu   // fenc(-inf)

// ---------------------------------------------------------------------------
// WMMA GEMM: C[M,N] = A[M,K] @ B[K,N], A bf16 row-major (lda=K),
// BT bf16 = B transposed, row-major [N,K] (ldb=K). K is a template param
// (64 / 128 / 224 / 39200), so the K-loop has a compile-time trip count.
// One wave computes a 32x32 tile (2x2 WMMA tiles): 4 independent WMMAs per
// K-step from 4 fragment loads. The steady-state loop loads the NEXT K-step
// unconditionally (final WMMA group peeled), so the double-buffer rotation
// coalesces into PHIs (no v_mov chains) and the wait before each WMMA group
// only covers the previous iteration's loads (partial s_wait_loadcnt ->
// VMEM/matrix overlap). Fragment layout per CDNA5 ISA 7.12.2:
//   lane<16: rows = lane, K-half 0; lane>=16: same rows, K-half 1.
//   elements 0..7  -> K = 8*half + 0..7
//   elements 8..15 -> K = 16 + 8*half + 0..7
// ---------------------------------------------------------------------------
DEVFN v16bf load_frag(const u16* row, int k0, int half) {
    union { uint4 u4[2]; v16bf v; } a;
    const u16* p = row + k0 + half * 8;
    a.u4[0] = *(const uint4*)(p);
    a.u4[1] = *(const uint4*)(p + 16);
    return a.v;
}

DEVFN v8f wmma_bf16(v16bf a, v16bf b, v8f c) {
    return __builtin_amdgcn_wmma_f32_16x16x32_bf16(
        false, a, false, b, (short)0, c, false, false);
}

template <int EPI>
DEVFN void store_tile(void* Cv, const float* bias, v8f acc,
                      int mbase, int n, int ldc) {
#pragma unroll
    for (int v = 0; v < 8; ++v) {
        int m = mbase + v;
        float x = acc[v];
        if (EPI == 2) {
            x += bias[n];
            ((float*)Cv)[(size_t)m * ldc + n] = 1.f / (1.f + __expf(-x));
        } else if (EPI == 1) {
            ((u16*)Cv)[(size_t)m * ldc + n] = f2bf(x);
        } else {
            ((float*)Cv)[(size_t)m * ldc + n] = x;
        }
    }
}

// EPI: 0 = store f32, 1 = store bf16(u16), 2 = +bias[n], sigmoid, store f32
template <int EPI, int KC>
__global__ __launch_bounds__(256) void k_gemm(
        const u16* __restrict__ A, const u16* __restrict__ BT,
        void* __restrict__ Cv, const float* __restrict__ bias,
        int M, int N, int ldc) {
    int lane = threadIdx.x & 31;
    long wave = (long)blockIdx.x * (blockDim.x >> 5) + (threadIdx.x >> 5);
    int mtiles = M >> 4, ntiles = N >> 4;
    int mblk = (mtiles + 1) >> 1, nblk = (ntiles + 1) >> 1;
    if (wave >= (long)mblk * nblk) return;
    int bm = (int)(wave / nblk), bn = (int)(wave % nblk);
    int m0 = bm << 5, n0 = bn << 5;            // 32x32 macro-tile origin
    bool vm1 = (m0 + 16) < M, vn1 = (n0 + 16) < N;
    int r = lane & 15, half = lane >> 4;

    // row pointers (clamped for odd trailing tiles; stores are guarded)
    int ar0 = m0 + r,      ar1 = vm1 ? m0 + 16 + r : M - 1;
    int br0 = n0 + r,      br1 = vn1 ? n0 + 16 + r : N - 1;
    const u16* arow0 = A  + (size_t)ar0 * KC;
    const u16* arow1 = A  + (size_t)ar1 * KC;
    const u16* brow0 = BT + (size_t)br0 * KC;
    const u16* brow1 = BT + (size_t)br1 * KC;

    v8f z = {0.f, 0.f, 0.f, 0.f, 0.f, 0.f, 0.f, 0.f};
    v8f acc00 = z, acc01 = z, acc10 = z, acc11 = z;

    // prologue: fragments for k0 = 0
    v16bf a0 = load_frag(arow0, 0, half);
    v16bf a1 = load_frag(arow1, 0, half);
    v16bf b0 = load_frag(brow0, 0, half);
    v16bf b1 = load_frag(brow1, 0, half);

    // steady state: unconditional next-loads, final WMMA group peeled below
#pragma unroll 2
    for (int k0 = 0; k0 + 32 < KC; k0 += 32) {
        v16bf na0 = load_frag(arow0, k0 + 32, half);
        v16bf na1 = load_frag(arow1, k0 + 32, half);
        v16bf nb0 = load_frag(brow0, k0 + 32, half);
        v16bf nb1 = load_frag(brow1, k0 + 32, half);
        if (KC > 256 && (k0 + 96 < KC)) {   // near-cache prefetch, long-K only
            __builtin_prefetch(arow0 + k0 + 96, 0, 3);
            __builtin_prefetch(arow1 + k0 + 96, 0, 3);
            __builtin_prefetch(brow0 + k0 + 96, 0, 3);
            __builtin_prefetch(brow1 + k0 + 96, 0, 3);
        }
        acc00 = wmma_bf16(a0, b0, acc00);
        acc01 = wmma_bf16(a0, b1, acc01);
        acc10 = wmma_bf16(a1, b0, acc10);
        acc11 = wmma_bf16(a1, b1, acc11);
        a0 = na0; a1 = na1; b0 = nb0; b1 = nb1;
    }
    // epilogue: last K-step
    acc00 = wmma_bf16(a0, b0, acc00);
    acc01 = wmma_bf16(a0, b1, acc01);
    acc10 = wmma_bf16(a1, b0, acc10);
    acc11 = wmma_bf16(a1, b1, acc11);

    int mb0 = m0 + (half << 3), mb1 = m0 + 16 + (half << 3);
    int nn0 = n0 + r, nn1 = n0 + 16 + r;
    store_tile<EPI>(Cv, bias, acc00, mb0, nn0, ldc);
    if (vn1)        store_tile<EPI>(Cv, bias, acc01, mb0, nn1, ldc);
    if (vm1)        store_tile<EPI>(Cv, bias, acc10, mb1, nn0, ldc);
    if (vm1 && vn1) store_tile<EPI>(Cv, bias, acc11, mb1, nn1, ldc);
}

// ---------------------------- elementwise kernels --------------------------
__global__ void k_fill_u32(u32* p, u32 v, long n) {
    long i = (long)blockIdx.x * blockDim.x + threadIdx.x;
    if (i < n) p[i] = v;
}

// fp32 [rows,cols] (row stride lds) -> bf16 [rows,kp], zero padded
__global__ void k_cvt_rm(const float* src, int lds, long rows, int cols,
                         u16* dst, int kp) {
    long i = (long)blockIdx.x * blockDim.x + threadIdx.x;
    if (i >= rows * kp) return;
    long r = i / kp; int c = (int)(i % kp);
    dst[i] = (c < cols) ? f2bf(src[r * (long)lds + c]) : (u16)0;
}

// weight W[K,N] (dense) -> BT bf16 [np, kp] with BT[n,k]=W[k,n], zero padded
__global__ void k_cvt_tr(const float* w, int K, int N, u16* dst, int kp, int np) {
    long i = (long)blockIdx.x * blockDim.x + threadIdx.x;
    if (i >= (long)np * kp) return;
    int n = (int)(i / kp), k = (int)(i % kp);
    dst[i] = (n < N && k < K) ? f2bf(w[(long)k * N + n]) : (u16)0;
}

// composition for loop branch: dst = bf16( x * (loop_rel + 1) ), padded
__global__ void k_comp(const float* x, int ldx, const float* lr,
                       long rows, int cols, u16* dst, int kp) {
    long i = (long)blockIdx.x * blockDim.x + threadIdx.x;
    if (i >= rows * kp) return;
    long r = i / kp; int c = (int)(i % kp);
    dst[i] = (c < cols) ? f2bf(x[r * (long)ldx + c] * (lr[c] + 1.f)) : (u16)0;
}

// gather + compose edge A-matrix: bf16( x[src[e]] * (rel[etype[e]] + 1) )
__global__ void k_edge_A(const float* x, int ldx, const float* rel, int ldrel,
                         const int* __restrict__ srcI, const int* __restrict__ etI,
                         long eh, int cols, u16* dst, int kp) {
    long i = (long)blockIdx.x * blockDim.x + threadIdx.x;
    if (i >= eh * kp) return;
    long e = i / kp; int c = (int)(i % kp);
    u16 v = 0;
    if (c < cols) {
        int s = srcI[e], t = etI[e];
        v = f2bf(x[(size_t)s * ldx + c] * (rel[(size_t)t * ldrel + c] + 1.f));
    }
    dst[i] = v;
}

// attention scores: sc[e,h] = leaky_relu( msg[e, h*dh : (h+1)*dh] . att[h] )
__global__ void k_attn(const u16* msg, int ldm, const float* att,
                       int H, int dh, long eh, float* sc) {
    long i = (long)blockIdx.x * blockDim.x + threadIdx.x;
    if (i >= eh * H) return;
    long e = i / H; int h = (int)(i % H);
    const u16* mrow = msg + (size_t)e * ldm + h * dh;
    const float* a = att + h * dh;
    float d = 0.f;
    for (int j = 0; j < dh; ++j) d += bf2f(mrow[j]) * a[j];
    sc[i] = (d > 0.f) ? d : 0.2f * d;    // leaky relu, slope 0.2
}

__global__ void k_segmax(const float* sc, const int* dstI, int H, long eh, u32* segmax) {
    long i = (long)blockIdx.x * blockDim.x + threadIdx.x;
    if (i >= eh * H) return;
    int node = dstI[i / H];
    atomicMax(&segmax[(size_t)node * H + (i % H)], fenc(sc[i]));
}

__global__ void k_expsum(float* sc, const int* dstI, int H, long eh,
                         const u32* segmax, float* segsum) {
    long i = (long)blockIdx.x * blockDim.x + threadIdx.x;
    if (i >= eh * H) return;
    size_t s = (size_t)dstI[i / H] * H + (i % H);
    float ex = __expf(sc[i] - fdec(segmax[s]));
    sc[i] = ex;
    atomicAdd(&segsum[s], ex);
}

__global__ void k_aggregate(const u16* msg, int ldm, const float* sc,
                            const float* segsum, const int* dstI,
                            int H, int dh, long eh, int cols, float* agg) {
    long i = (long)blockIdx.x * blockDim.x + threadIdx.x;
    if (i >= eh * cols) return;
    long e = i / cols; int d = (int)(i % cols);
    int h = d / dh;
    int node = dstI[e];
    size_t s = (size_t)node * H + h;
    float alpha = sc[e * H + h] / (segsum[s] + 1e-16f);
    atomicAdd(&agg[(size_t)node * P200 + d], bf2f(msg[(size_t)e * ldm + d]) * alpha);
}

__global__ void k_combine(float* agg, const float* loopC, long n) {
    long i = (long)blockIdx.x * blockDim.x + threadIdx.x;
    if (i < n) agg[i] = (agg[i] + loopC[i]) * (1.f / 3.f);
}

// per-column mean / rsqrt(var+eps) over M rows (blockIdx.x = column)
__global__ void k_col_stats(const float* X, int ld, long M, float* mean, float* rsig) {
    __shared__ float ss[256], sq[256];
    int c = blockIdx.x, t = threadIdx.x;
    float s = 0.f, q = 0.f;
    for (long r = t; r < M; r += 256) {
        float v = X[r * (long)ld + c];
        s += v; q += v * v;
    }
    ss[t] = s; sq[t] = q; __syncthreads();
    for (int w = 128; w > 0; w >>= 1) {
        if (t < w) { ss[t] += ss[t + w]; sq[t] += sq[t + w]; }
        __syncthreads();
    }
    if (t == 0) {
        float m = ss[0] / (float)M;
        float v = sq[0] / (float)M - m * m;
        mean[c] = m; rsig[c] = rsqrtf(v + EPS_BN);
    }
}

// encoder output: tanh(BN(x)); write f32 + bf16 copies (ld = P200)
__global__ void k_bn_tanh(const float* X, long M, int cols,
                          const float* mean, const float* rsig,
                          float* of, u16* ob) {
    long i = (long)blockIdx.x * blockDim.x + threadIdx.x;
    if (i >= M * P200) return;
    int d = (int)(i % P200);
    float y = 0.f;
    if (d < cols) y = tanhf((X[i] - mean[d]) * rsig[d]);
    of[i] = y; ob[i] = f2bf(y);
}

// gather + interleave into ConvE input image: stk[b][2d]=sub, [2d+1]=rel
__global__ void k_stk(const float* ent2f, const float* rel2f, int ld,
                      const int* sub, const int* relI, float* stk) {
    long i = (long)blockIdx.x * blockDim.x + threadIdx.x;
    if (i >= (long)BSZ * 200) return;
    int b = (int)(i / 200), d = (int)(i % 200);
    stk[(size_t)b * 400 + 2 * d]     = ent2f[(size_t)sub[b] * ld + d];
    stk[(size_t)b * 400 + 2 * d + 1] = rel2f[(size_t)relI[b] * ld + d];
}

// global mean / rsig over whole image batch (single block)
__global__ void k_gstats(const float* X, long n, float* gs) {
    __shared__ float ss[256], sq[256];
    int t = threadIdx.x;
    float s = 0.f, q = 0.f;
    for (long i = t; i < n; i += 256) { float v = X[i]; s += v; q += v * v; }
    ss[t] = s; sq[t] = q; __syncthreads();
    for (int w = 128; w > 0; w >>= 1) {
        if (t < w) { ss[t] += ss[t + w]; sq[t] += sq[t + w]; }
        __syncthreads();
    }
    if (t == 0) {
        float m = ss[0] / (float)n;
        float v = sq[0] / (float)n - m * m;
        gs[0] = m; gs[1] = rsqrtf(v + EPS_BN);
    }
}

// im2col of BN-normalized 20x20 image; A[(b*196+p), kk], kk in [0,64)
__global__ void k_im2col(const float* stk, const float* gs, u16* dst) {
    long i = (long)blockIdx.x * blockDim.x + threadIdx.x;
    if (i >= (long)BSZ * 196 * KCONV) return;
    int kk = (int)(i & 63);
    long bp = i >> 6;
    int p = (int)(bp % 196), b = (int)(bp / 196);
    u16 v = 0;
    if (kk < 49) {
        int ky = kk / 7, kx = kk % 7, oy = p / 14, ox = p % 14;
        float x = stk[(size_t)b * 400 + (oy + ky) * 20 + (ox + kx)];
        v = f2bf((x - gs[0]) * gs[1]);
    }
    dst[i] = v;
}

__global__ void k_convwT(const float* cw, u16* dst) {
    long i = (long)blockIdx.x * blockDim.x + threadIdx.x;
    if (i >= (long)PCONV * KCONV) return;
    int f = (int)(i / KCONV), kk = (int)(i % KCONV);
    dst[i] = (f < 200 && kk < 49) ? f2bf(cw[(long)f * 49 + kk]) : (u16)0;
}

// BN(per-channel)+relu on conv output, repack to fc layout [b, c*196+p] bf16
__global__ void k_conv_pack(const float* convC, const float* mean, const float* rsig,
                            u16* fcA) {
    long i = (long)blockIdx.x * blockDim.x + threadIdx.x;
    if (i >= (long)BSZ * 200 * 196) return;
    int p = (int)(i % 196);
    int c = (int)((i / 196) % 200);
    int b = (int)(i / (196 * 200));
    float x = convC[((size_t)b * 196 + p) * PCONV + c];
    float y = (x - mean[c]) * rsig[c];
    if (y < 0.f) y = 0.f;
    fcA[(size_t)b * FLATSZ + (size_t)c * 196 + p] = f2bf(y);
}

__global__ void k_fcwT(const float* fw, u16* dst) {
    long i = (long)blockIdx.x * blockDim.x + threadIdx.x;
    if (i >= (long)P200 * FLATSZ) return;
    int n = (int)(i / FLATSZ);
    long k = i % FLATSZ;
    dst[i] = (n < 200) ? f2bf(fw[k * 200 + n]) : (u16)0;
}

__global__ void k_bn_relu_b(const float* hC, const float* mean, const float* rsig,
                            u16* hbf) {
    long i = (long)blockIdx.x * blockDim.x + threadIdx.x;
    if (i >= (long)BSZ * P200) return;
    int d = (int)(i % P200);
    float y = 0.f;
    if (d < 200) {
        y = (hC[i] - mean[d]) * rsig[d];
        if (y < 0.f) y = 0.f;
    }
    hbf[i] = f2bf(y);
}

// ------------------------------- host side ---------------------------------
static inline unsigned blks(long n) { return (unsigned)((n + 255) / 256); }

template <int EPI>
static void launch_gemm_k(hipStream_t st, const u16* A, const u16* BT, void* C,
                          const float* bias, int M, int N, int K, int ldc,
                          unsigned nb) {
    switch (K) {
    case KCONV:  k_gemm<EPI, KCONV> <<<nb, 256, 0, st>>>(A, BT, C, bias, M, N, ldc); break;
    case P100:   k_gemm<EPI, P100>  <<<nb, 256, 0, st>>>(A, BT, C, bias, M, N, ldc); break;
    case P200:   k_gemm<EPI, P200>  <<<nb, 256, 0, st>>>(A, BT, C, bias, M, N, ldc); break;
    default:     k_gemm<EPI, FLATSZ><<<nb, 256, 0, st>>>(A, BT, C, bias, M, N, ldc); break;
    }
}

// lda == ldb == K for every call site by construction.
static void run_gemm(hipStream_t st, const u16* A, const u16* BT, void* C,
                     const float* bias, int M, int N, int K, int ldc, int epi) {
    int mblk = ((M >> 4) + 1) >> 1, nblk = ((N >> 4) + 1) >> 1;
    long tiles = (long)mblk * nblk;
    unsigned nb = (unsigned)((tiles + 7) / 8);   // 8 waves / block (256 thr)
    switch (epi) {
    case 0:  launch_gemm_k<0>(st, A, BT, C, bias, M, N, K, ldc, nb); break;
    case 1:  launch_gemm_k<1>(st, A, BT, C, bias, M, N, K, ldc, nb); break;
    default: launch_gemm_k<2>(st, A, BT, C, bias, M, N, K, ldc, nb); break;
    }
}

extern "C" void kernel_launch(void* const* d_in, const int* in_sizes, int n_in,
                              void* d_out, int out_size, void* d_ws, size_t ws_size,
                              hipStream_t stream) {
    (void)in_sizes; (void)n_in; (void)out_size; (void)ws_size;

    const float* init_embed = (const float*)d_in[0];
    const float* init_rel   = (const float*)d_in[1];
    const float* loop_rel1  = (const float*)d_in[2];
    const float* w_in1      = (const float*)d_in[3];
    const float* w_out1     = (const float*)d_in[4];
    const float* w_loop1    = (const float*)d_in[5];
    const float* w_rel1     = (const float*)d_in[6];
    const float* att1       = (const float*)d_in[7];
    const float* loop_rel2  = (const float*)d_in[8];
    const float* w_in2      = (const float*)d_in[9];
    const float* w_out2     = (const float*)d_in[10];
    const float* w_loop2    = (const float*)d_in[11];
    const float* w_rel2     = (const float*)d_in[12];
    const float* att2       = (const float*)d_in[13];
    const float* conv_w     = (const float*)d_in[14];
    const float* fc_w       = (const float*)d_in[15];
    // fc_b (d_in[16]) provably cancels through the following axis-0 batchnorm.
    const float* ent_bias   = (const float*)d_in[17];
    const int*   sub        = (const int*)d_in[18];
    const int*   relI       = (const int*)d_in[19];
    const int*   edge_index = (const int*)d_in[20];
    const int*   edge_type  = (const int*)d_in[21];
    float* out = (float*)d_out;

    // ---- workspace layout (region reuse; total ~540 MB) ----
    size_t off = 0;
    auto alloc = [&](size_t bytes) -> char* {
        char* p = (char*)d_ws + off;
        off = (off + bytes + 255) & ~(size_t)255;
        return p;
    };
    u16*  bufA    = (u16*)alloc((size_t)E_HALF * P200 * 2);     // edge A / im2col / fcA
    char* region2 = alloc((size_t)BSZ * 196 * PCONV * 4);       // msg bf16 / convC f32
    u16*  msg     = (u16*)region2;
    float* convC  = (float*)region2;
    float* ent1f  = (float*)alloc((size_t)NUM_ENT * P200 * 4);
    float* ent2f  = (float*)alloc((size_t)NUM_ENT * P200 * 4);
    u16*  ent1bf  = (u16*)alloc((size_t)NUM_ENT * P200 * 2);
    u16*  ent2bf  = (u16*)alloc((size_t)NUM_ENT * P200 * 2);
    float* agg    = (float*)alloc((size_t)NUM_ENT * P200 * 4);
    float* loopC  = (float*)alloc((size_t)NUM_ENT * P200 * 4);
    float* sc     = (float*)alloc((size_t)E_HALF * 2 * 4);
    u32*  segmax  = (u32*)alloc((size_t)NUM_ENT * 2 * 4);
    float* segsum = (float*)alloc((size_t)NUM_ENT * 2 * 4);
    float* rel1f  = (float*)alloc((size_t)NUM_REL2 * P200 * 4);
    float* rel2f  = (float*)alloc((size_t)NUM_REL2 * P200 * 4);
    u16*  relA    = (u16*)alloc((size_t)NUM_REL2 * P200 * 2);
    u16*  wInT    = (u16*)alloc((size_t)P200 * P200 * 2);
    u16*  wOutT   = (u16*)alloc((size_t)P200 * P200 * 2);
    u16*  wLoopT  = (u16*)alloc((size_t)P200 * P200 * 2);
    u16*  wRelT   = (u16*)alloc((size_t)P200 * P200 * 2);
    u16*  xcomp   = (u16*)alloc((size_t)NUM_ENT * P200 * 2);
    u16*  convwT  = (u16*)alloc((size_t)PCONV * KCONV * 2);
    u16*  fcwT    = (u16*)alloc((size_t)P200 * FLATSZ * 2);
    float* stk    = (float*)alloc((size_t)BSZ * 400 * 4);
    float* statM  = (float*)alloc(P200 * 4);
    float* statR  = (float*)alloc(P200 * 4);
    float* gstats = (float*)alloc(8 * 4);
    float* hC     = (float*)alloc((size_t)BSZ * P200 * 4);
    u16*  hbf     = (u16*)alloc((size_t)BSZ * P200 * 2);

    // ------------------- one RAGAT layer -------------------
    auto run_layer = [&](const float* x, int ldx, int K, int Kp,
                         const float* relIn, int ldrel, const float* loopRel,
                         const float* wIn, const float* wOut,
                         const float* wLoop, const float* wRel,
                         const float* att, int H,
                         float* entf, u16* entbf, float* relOutF) {
        int dh = 200 / H;
        // weight transposes -> bf16 [P200, Kp]
        k_cvt_tr<<<blks((long)P200 * Kp), 256, 0, stream>>>(wIn,   K, 200, wInT,   Kp, P200);
        k_cvt_tr<<<blks((long)P200 * Kp), 256, 0, stream>>>(wOut,  K, 200, wOutT,  Kp, P200);
        k_cvt_tr<<<blks((long)P200 * Kp), 256, 0, stream>>>(wLoop, K, 200, wLoopT, Kp, P200);
        k_cvt_tr<<<blks((long)P200 * Kp), 256, 0, stream>>>(wRel,  K, 200, wRelT,  Kp, P200);
        // loop branch: comp(x, loop_rel) @ w_loop
        k_comp<<<blks((long)NUM_ENT * Kp), 256, 0, stream>>>(x, ldx, loopRel, NUM_ENT, K, xcomp, Kp);
        run_gemm(stream, xcomp, wLoopT, loopC, nullptr, NUM_ENT, P200, Kp, P200, 0);
        // rel transform: rel @ w_rel
        k_cvt_rm<<<blks((long)NUM_REL2 * Kp), 256, 0, stream>>>(relIn, ldrel, NUM_REL2, K, relA, Kp);
        run_gemm(stream, relA, wRelT, relOutF, nullptr, NUM_REL2, P200, Kp, P200, 0);
        // zero aggregation accumulator
        k_fill_u32<<<blks((long)NUM_ENT * P200), 256, 0, stream>>>((u32*)agg, 0u, (long)NUM_ENT * P200);

        for (int dir = 0; dir < 2; ++dir) {
            const int* dstI = edge_index + (dir ? E_HALF : 0);
            const int* srcI = edge_index + E_TOT + (dir ? E_HALF : 0);
            const int* etI  = edge_type + (dir ? E_HALF : 0);
            const u16* Wt = dir ? wOutT : wInT;
            k_edge_A<<<blks((long)E_HALF * Kp), 256, 0, stream>>>(
                x, ldx, relIn, ldrel, srcI, etI, E_HALF, K, bufA, Kp);
            run_gemm(stream, bufA, Wt, msg, nullptr, E_HALF, P200, Kp, P200, 1);
            k_attn<<<blks((long)E_HALF * H), 256, 0, stream>>>(msg, P200, att, H, dh, E_HALF, sc);
            k_fill_u32<<<blks((long)NUM_ENT * H), 256, 0, stream>>>(segmax, ENC_NEG_INF, (long)NUM_ENT * H);
            k_fill_u32<<<blks((long)NUM_ENT * H), 256, 0, stream>>>((u32*)segsum, 0u, (long)NUM_ENT * H);
            k_segmax<<<blks((long)E_HALF * H), 256, 0, stream>>>(sc, dstI, H, E_HALF, segmax);
            k_expsum<<<blks((long)E_HALF * H), 256, 0, stream>>>(sc, dstI, H, E_HALF, segmax, segsum);
            k_aggregate<<<blks((long)E_HALF * 200), 256, 0, stream>>>(
                msg, P200, sc, segsum, dstI, H, dh, E_HALF, 200, agg);
        }
        k_combine<<<blks((long)NUM_ENT * P200), 256, 0, stream>>>(agg, loopC, (long)NUM_ENT * P200);
        k_col_stats<<<200, 256, 0, stream>>>(agg, P200, NUM_ENT, statM, statR);
        k_bn_tanh<<<blks((long)NUM_ENT * P200), 256, 0, stream>>>(
            agg, NUM_ENT, 200, statM, statR, entf, entbf);
    };

    // ------------------- encoder -------------------
    run_layer(init_embed, 100, 100, P100, init_rel, 100, loop_rel1,
              w_in1, w_out1, w_loop1, w_rel1, att1, 2, ent1f, ent1bf, rel1f);
    run_layer(ent1f, P200, 200, P200, rel1f, P200, loop_rel2,
              w_in2, w_out2, w_loop2, w_rel2, att2, 1, ent2f, ent2bf, rel2f);

    // ------------------- ConvE decoder -------------------
    k_stk<<<blks((long)BSZ * 200), 256, 0, stream>>>(ent2f, rel2f, P200, sub, relI, stk);
    k_gstats<<<1, 256, 0, stream>>>(stk, (long)BSZ * 400, gstats);
    k_im2col<<<blks((long)BSZ * 196 * KCONV), 256, 0, stream>>>(stk, gstats, bufA);
    k_convwT<<<blks((long)PCONV * KCONV), 256, 0, stream>>>(conv_w, convwT);
    run_gemm(stream, bufA, convwT, convC, nullptr, BSZ * 196, PCONV, KCONV, PCONV, 0);
    k_col_stats<<<200, 256, 0, stream>>>(convC, PCONV, (long)BSZ * 196, statM, statR);
    k_conv_pack<<<blks((long)BSZ * 200 * 196), 256, 0, stream>>>(convC, statM, statR, bufA);
    k_fcwT<<<blks((long)P200 * FLATSZ), 256, 0, stream>>>(fc_w, fcwT);
    run_gemm(stream, bufA, fcwT, hC, nullptr, BSZ, P200, FLATSZ, P200, 0);
    k_col_stats<<<200, 256, 0, stream>>>(hC, P200, BSZ, statM, statR);
    k_bn_relu_b<<<blks((long)BSZ * P200), 256, 0, stream>>>(hC, statM, statR, hbf);
    // logits = h @ ent2.T + ent_bias, sigmoid  (BT = ent2bf itself, [40000, 224])
    run_gemm(stream, hbf, ent2bf, out, ent_bias, BSZ, NUM_ENT, P200, NUM_ENT, 2);
}
